// Transducer_30116310679842
// MI455X (gfx1250) — matrix operand order repaired and checked by reading, not compiled, not run on hardware
//
#include <hip/hip_runtime.h>
#include <hip/hip_bf16.h>
#include <math.h>

// ---------------------------------------------------------------------------
// RNN-T (Transducer) forward loss for MI455X (gfx1250, wave32, WMMA).
// All matmuls run through v_wmma_f32_16x16x32_bf16 (bf16 in, f32 accumulate).
// Fragment loads are vectorized 16-byte chunks (global_load_b128 / ds_load_b128).
// ---------------------------------------------------------------------------

typedef __bf16 bf16_t;
typedef __attribute__((ext_vector_type(16))) __bf16 bf16x16;
typedef __attribute__((ext_vector_type(8)))  __bf16 bf16x8;
typedef __attribute__((ext_vector_type(8)))  float  floatx8;

__device__ __forceinline__ floatx8 wmma_bf16f32(bf16x16 a, bf16x16 b, floatx8 c) {
  // 8 args: (neg_a, A, neg_b, B, c_mod, C, reuse_a, reuse_b)
  return __builtin_amdgcn_wmma_f32_16x16x32_bf16(false, a, false, b, (short)0, c,
                                                 false, false);
}

__device__ __forceinline__ bf16x16 concat8(bf16x8 a, bf16x8 b) {
  return __builtin_shufflevector(a, b, 0, 1, 2, 3, 4, 5, 6, 7,
                                 8, 9, 10, 11, 12, 13, 14, 15);
}

__device__ __forceinline__ float sigmoidf_(float x) { return 1.0f / (1.0f + expf(-x)); }

__device__ __forceinline__ float logaddexp_f(float x, float y) {
  float m = fmaxf(x, y);
  float d = fminf(x, y) - m;
  return m + log1pf(expf(d));
}

// --- WMMA fragment loaders (ISA 7.12.2 layouts) -----------------------------
// A 16x32 bf16 (row-major [.., lda]): lane&15 = row; lanes 0-15 take K
// {0..7, 16..23}, lanes 16-31 take K {8..15, 24..31}.  Both halves are
// contiguous 8xbf16 chunks -> two 16-byte vector loads.
__device__ __forceinline__ bf16x16 load_a_fast(const bf16_t* __restrict__ A, int lda,
                                               int m0, int k0, int lane) {
  int row   = m0 + (lane & 15);
  int khalf = (lane >> 4) * 8;
  const bf16_t* ap = A + (size_t)row * lda + k0 + khalf;
  bf16x8 lo = *(const bf16x8*)(ap);
  bf16x8 hi = *(const bf16x8*)(ap + 16);
  return concat8(lo, hi);
}

// B 32x16 where B(k,n) = W[n,k], W row-major [N, ldw]: lane&15 = n;
// lanes 0-15 take K {0..7, 16..23}, lanes 16-31 take K {8..15, 24..31}.
__device__ __forceinline__ bf16x16 load_b_fast(const bf16_t* __restrict__ W, int ldw,
                                               int n0, int k0, int lane) {
  int n  = n0 + (lane & 15);
  int kb = k0 + (lane >> 4) * 8;
  const bf16_t* wp = W + (size_t)n * ldw + kb;
  bf16x8 lo = *(const bf16x8*)(wp);
  bf16x8 hi = *(const bf16x8*)(wp + 16);
  return concat8(lo, hi);
}

// Guarded chunk: vector load on the fast path, element loads only at the edge.
__device__ __forceinline__ bf16x8 load_chunk_guard(const bf16_t* __restrict__ base,
                                                   int kk, int K, bool rok) {
  bf16x8 v;
  if (rok && (kk + 8 <= K)) {
    v = *(const bf16x8*)(base + kk);
  } else {
#pragma unroll
    for (int i = 0; i < 8; ++i)
      v[i] = (rok && (kk + i < K)) ? base[kk + i] : (bf16_t)0.0f;
  }
  return v;
}

__device__ __forceinline__ bf16x16 load_a_guard(const bf16_t* __restrict__ A, int lda,
                                                int m0, int k0, int M, int K, int lane) {
  int row   = m0 + (lane & 15);
  int khalf = (lane >> 4) * 8;
  bool rok  = row < M;
  const bf16_t* ap = A + (size_t)row * lda;
  bf16x8 lo = load_chunk_guard(ap, k0 + khalf, K, rok);
  bf16x8 hi = load_chunk_guard(ap, k0 + khalf + 16, K, rok);
  return concat8(lo, hi);
}

__device__ __forceinline__ bf16x16 load_b_guard(const bf16_t* __restrict__ W, int ldw,
                                                int n0, int k0, int N, int K, int lane) {
  int n  = n0 + (lane & 15);
  int kb = k0 + (lane >> 4) * 8;
  bool nok = n < N;
  const bf16_t* wp = W + (size_t)n * ldw;
  bf16x8 lo = load_chunk_guard(wp, kb, K, nok);
  bf16x8 hi = load_chunk_guard(wp, kb + 16, K, nok);
  return concat8(lo, hi);
}

// ---------------------------------------------------------------------------
// Generic WMMA GEMM: C[M,N] = A[M,K] @ W[N,K]^T + bias0 + bias1
// One 16x16 tile per wave; 256-thread blocks = 8 waves.
// GUARD=false requires M%16==0, N%16==0, K%32==0 (branch-free inner loop).
// ---------------------------------------------------------------------------
template <bool GUARD>
__global__ void gemm_bf16_wmma(const bf16_t* __restrict__ A, int lda,
                               const bf16_t* __restrict__ W, int ldw,
                               const float* __restrict__ bias0,
                               const float* __restrict__ bias1,
                               float* __restrict__ Cf, bf16_t* __restrict__ Cb,
                               int ldc, int M, int N, int K) {
  int lane = threadIdx.x & 31;
  int wave = threadIdx.x >> 5;
  int ntN = (N + 15) >> 4;
  int ntM = (M + 15) >> 4;
  int tile = blockIdx.x * (blockDim.x >> 5) + wave;
  if (tile >= ntM * ntN) return;
  int m0 = (tile / ntN) * 16;
  int n0 = (tile % ntN) * 16;

  floatx8 c;
#pragma unroll
  for (int j = 0; j < 8; ++j) c[j] = 0.0f;

  for (int k0 = 0; k0 < K; k0 += 32) {
    bf16x16 af, bf;
    if (GUARD) {
      af = load_a_guard(A, lda, m0, k0, M, K, lane);
      bf = load_b_guard(W, ldw, n0, k0, N, K, lane);
    } else {
      af = load_a_fast(A, lda, m0, k0, lane);
      bf = load_b_fast(W, ldw, n0, k0, lane);
    }
    c = wmma_bf16f32(af, bf, c);
  }

  int n = n0 + (lane & 15);
  if (!GUARD || n < N) {
    float bias = (bias0 ? bias0[n] : 0.0f) + (bias1 ? bias1[n] : 0.0f);
    int mb = (lane >> 4) * 8;
#pragma unroll
    for (int j = 0; j < 8; ++j) {
      int m = m0 + mb + j;
      if (!GUARD || m < M) {
        float v = c[j] + bias;
        if (Cf) Cf[(size_t)m * ldc + n] = v;
        if (Cb) Cb[(size_t)m * ldc + n] = (bf16_t)v;
      }
    }
  }
}

// ---------------------------------------------------------------------------
// One LSTM timestep: z = gates_in[:,t,:] + h_prev @ Whh^T ; apply cell.
// Grid = H/16 blocks, 128 threads (4 waves = 4 gates i,f,g,o per 16 units).
// h buffers are [16, H] bf16 with rows 4..15 pre-zeroed (batch pad for WMMA).
// ---------------------------------------------------------------------------
__global__ void lstm_step_kernel(const bf16_t* __restrict__ hprev,
                                 bf16_t* __restrict__ hnext,
                                 const bf16_t* __restrict__ Whh,  // [4H, H]
                                 const float* __restrict__ gates, // [B*T, 4H]
                                 float* __restrict__ cbuf,        // [4, H]
                                 float* __restrict__ hs_f,        // [B*T, H]
                                 bf16_t* __restrict__ hs_b,       // [B*T, H]
                                 int H, int T, int t) {
  __shared__ float zsh[4][4][16];
  int lane = threadIdx.x & 31;
  int wave = threadIdx.x >> 5;  // gate index: 0=i 1=f 2=g 3=o
  int n0 = blockIdx.x * 16;

  floatx8 c;
#pragma unroll
  for (int j = 0; j < 8; ++j) c[j] = 0.0f;

  const bf16_t* Wg = Whh + (size_t)(wave * H + n0) * H;
  for (int k0 = 0; k0 < H; k0 += 32) {
    bf16x16 af = load_a_fast(hprev, H, 0, k0, lane);
    bf16x16 bf = load_b_fast(Wg, H, 0, k0, lane);
    c = wmma_bf16f32(af, bf, c);
  }

  if (lane < 16) {  // lanes 0-15 hold rows M=0..7 (VGPR j -> M=j); batch = rows 0..3
#pragma unroll
    for (int j = 0; j < 4; ++j) {
      float g = gates[(size_t)(j * T + t) * (size_t)(4 * H) + (size_t)(wave * H + n0 + lane)];
      zsh[wave][j][lane] = c[j] + g;
    }
  }
  __syncthreads();

  int tid = threadIdx.x;
  if (tid < 64) {
    int b = tid >> 4, n = tid & 15;
    float zi = zsh[0][b][n], zf = zsh[1][b][n], zg = zsh[2][b][n], zo = zsh[3][b][n];
    int col = n0 + n;
    float cold = cbuf[b * H + col];
    float cn = sigmoidf_(zf) * cold + sigmoidf_(zi) * tanhf(zg);
    float h  = sigmoidf_(zo) * tanhf(cn);
    cbuf[b * H + col]  = cn;
    hnext[b * H + col] = (bf16_t)h;
    size_t idx = (size_t)(b * T + t) * H + col;
    hs_f[idx] = h;
    hs_b[idx] = (bf16_t)h;
  }
}

// ---------------------------------------------------------------------------
// LayerNorm over rows of X [R, D]; emits f32 and/or bf16.
// ---------------------------------------------------------------------------
__global__ void layernorm_kernel(const float* __restrict__ X,
                                 const float* __restrict__ gam,
                                 const float* __restrict__ bet,
                                 float* __restrict__ Yf, bf16_t* __restrict__ Yb,
                                 int R, int D) {
  int r = blockIdx.x;
  if (r >= R) return;
  const float* x = X + (size_t)r * D;
  __shared__ float s1[256], s2[256];
  float a = 0.0f, b = 0.0f;
  for (int i = threadIdx.x; i < D; i += blockDim.x) {
    float v = x[i]; a += v; b += v * v;
  }
  s1[threadIdx.x] = a; s2[threadIdx.x] = b;
  __syncthreads();
  for (int off = 128; off > 0; off >>= 1) {
    if ((int)threadIdx.x < off) { s1[threadIdx.x] += s1[threadIdx.x + off];
                                  s2[threadIdx.x] += s2[threadIdx.x + off]; }
    __syncthreads();
  }
  float mu  = s1[0] / (float)D;
  float var = s2[0] / (float)D - mu * mu;
  float inv = rsqrtf(var + 1e-5f);
  for (int i = threadIdx.x; i < D; i += blockDim.x) {
    float v = (x[i] - mu) * inv * gam[i] + bet[i];
    if (Yf) Yf[(size_t)r * D + i] = v;
    if (Yb) Yb[(size_t)r * D + i] = (bf16_t)v;
  }
}

__global__ void add_inplace_kernel(float* __restrict__ y, const float* __restrict__ x,
                                   size_t n) {
  size_t i = (size_t)blockIdx.x * blockDim.x + threadIdx.x;
  if (i < n) y[i] += x[i];
}

__global__ void time_reduce_kernel(const float* __restrict__ x, float* __restrict__ yf,
                                   bf16_t* __restrict__ yb, int B, int Tin, int H) {
  int Tout = Tin / 2;
  size_t total = (size_t)B * Tout * H;
  size_t i = (size_t)blockIdx.x * blockDim.x + threadIdx.x;
  if (i >= total) return;
  int h = (int)(i % H);
  size_t rem = i / H;
  int t2 = (int)(rem % Tout);
  int b  = (int)(rem / Tout);
  size_t i0 = ((size_t)b * Tin + 2 * t2) * H + h;
  float v = 0.5f * (x[i0] + x[i0 + H]);
  yf[i] = v;
  yb[i] = (bf16_t)v;
}

__global__ void f32_to_bf16_kernel(const float* __restrict__ x, bf16_t* __restrict__ y,
                                   size_t n) {
  size_t i = (size_t)blockIdx.x * blockDim.x + threadIdx.x;
  if (i < n) y[i] = (bf16_t)x[i];
}

__global__ void zero_f32_kernel(float* p, size_t n) {
  size_t i = (size_t)blockIdx.x * blockDim.x + threadIdx.x;
  if (i < n) p[i] = 0.0f;
}
__global__ void zero_b16_kernel(bf16_t* p, size_t n) {
  size_t i = (size_t)blockIdx.x * blockDim.x + threadIdx.x;
  if (i < n) p[i] = (bf16_t)0.0f;
}

// Decoder embedding with prepended BOS (=0); table row 0 is the zeroed pad row.
__global__ void embed_kernel(const int* __restrict__ ys, const float* __restrict__ table,
                             bf16_t* __restrict__ out, int B, int U1, int E) {
  size_t total = (size_t)B * U1 * E;
  size_t i = (size_t)blockIdx.x * blockDim.x + threadIdx.x;
  if (i >= total) return;
  int e = (int)(i % E);
  size_t rem = i / E;
  int u = (int)(rem % U1);
  int b = (int)(rem / U1);
  int tok = (u == 0) ? 0 : ys[b * (U1 - 1) + (u - 1)];
  out[i] = (bf16_t)table[(size_t)tok * E + e];
}

// ---------------------------------------------------------------------------
// Fused joint: h = tanh(pe[b,t] + pd[b,u] + b1) ; logits = h @ W2^T + b2 ;
// log-softmax over V ; emit lp_blank[b,t,u] and lp_y[b,t,u] (label-gathered,
// ylen-masked). One block per 16 (b,t,u) rows; logits never hit HBM.
// ---------------------------------------------------------------------------
__global__ void joint_softmax_kernel(const float* __restrict__ pe,   // [B*Tp, JH]
                                     const float* __restrict__ pd,   // [B*U1, JH]
                                     const float* __restrict__ b1,   // [JH]
                                     const bf16_t* __restrict__ W2,  // [V, JH]
                                     const float* __restrict__ b2,   // [V]
                                     const int* __restrict__ ys,     // [B, U]
                                     const int* __restrict__ ylen,   // [B]
                                     float* __restrict__ lp_blank,   // [B*Tp*U1]
                                     float* __restrict__ lp_y,       // [B*Tp*U]
                                     int B, int Tp, int U1, int JH, int V) {
  extern __shared__ unsigned char smem[];
  bf16_t* hsh = (bf16_t*)smem;                                     // [16][JH]
  float*  lsh = (float*)(smem + (size_t)16 * JH * sizeof(bf16_t)); // [16][V]
  float*  red = lsh + (size_t)16 * V;                              // [16*16]
  float*  rowm = red + 256;                                        // [16]
  float*  rowl = rowm + 16;                                        // [16]

  int U = U1 - 1;
  int R = B * Tp * U1;
  int m0 = blockIdx.x * 16;

  // 1. Build tanh hidden rows (bf16) in LDS.
  for (int idx = threadIdx.x; idx < 16 * JH; idx += blockDim.x) {
    int row = idx / JH, col = idx % JH;
    int gr = m0 + row;
    float v = 0.0f;
    if (gr < R) {
      int u  = gr % U1;
      int bt = gr / U1;           // b*Tp + t
      int b  = bt / Tp;
      v = tanhf(pe[(size_t)bt * JH + col] + pd[(size_t)(b * U1 + u) * JH + col] + b1[col]);
    }
    hsh[idx] = (bf16_t)v;
  }
  __syncthreads();

  // 2. WMMA: 16 x V logits into LDS (8 waves x 8 N-tiles covers V=1024).
  int lane = threadIdx.x & 31;
  int wave = threadIdx.x >> 5;
  int ntN = V >> 4;
  int nw  = blockDim.x >> 5;
  for (int nt = wave; nt < ntN; nt += nw) {
    int n0 = nt * 16;
    floatx8 c;
#pragma unroll
    for (int j = 0; j < 8; ++j) c[j] = 0.0f;
    for (int k0 = 0; k0 < JH; k0 += 32) {
      bf16x16 af = load_a_fast(hsh, JH, 0, k0, lane);   // ds_load_b128 x2
      bf16x16 bf = load_b_fast(W2, JH, n0, k0, lane);   // global_load_b128 x2
      c = wmma_bf16f32(af, bf, c);
    }
    int n = n0 + (lane & 15);
    int mb = (lane >> 4) * 8;
    float bias = b2[n];
#pragma unroll
    for (int j = 0; j < 8; ++j) lsh[(size_t)(mb + j) * V + n] = c[j] + bias;
  }
  __syncthreads();

  // 3. Row-wise max / logsumexp (16 threads per row).
  int row = threadIdx.x >> 4, seg = threadIdx.x & 15;
  {
    float m = -3.4e38f;
    for (int n = seg; n < V; n += 16) m = fmaxf(m, lsh[(size_t)row * V + n]);
    red[row * 16 + seg] = m;
  }
  __syncthreads();
  if (threadIdx.x < 16) {
    float m = red[threadIdx.x * 16];
    for (int i = 1; i < 16; ++i) m = fmaxf(m, red[threadIdx.x * 16 + i]);
    rowm[threadIdx.x] = m;
  }
  __syncthreads();
  {
    float m = rowm[row], s = 0.0f;
    for (int n = seg; n < V; n += 16) s += expf(lsh[(size_t)row * V + n] - m);
    red[row * 16 + seg] = s;
  }
  __syncthreads();
  if (threadIdx.x < 16) {
    float s = 0.0f;
    for (int i = 0; i < 16; ++i) s += red[threadIdx.x * 16 + i];
    rowl[threadIdx.x] = rowm[threadIdx.x] + logf(s);
  }
  __syncthreads();

  // 4. Emit lp_blank / lp_y.
  if (threadIdx.x < 16) {
    int gr = m0 + (int)threadIdx.x;
    if (gr < R) {
      int u  = gr % U1;
      int bt = gr / U1;
      int b  = bt / Tp;
      float lse = rowl[threadIdx.x];
      lp_blank[gr] = lsh[(size_t)threadIdx.x * V + 0] - lse;
      if (u < U) {
        int label = ys[b * U + u];
        float v = (u < ylen[b]) ? (lsh[(size_t)threadIdx.x * V + label] - lse) : -1e30f;
        lp_y[(size_t)bt * U + u] = v;
      }
    }
  }
}

// ---------------------------------------------------------------------------
// RNN-T alpha recursion (tiny, sequential): one lane per batch element.
// ---------------------------------------------------------------------------
__global__ void rnnt_alpha_kernel(const float* __restrict__ lp_blank,
                                  const float* __restrict__ lp_y,
                                  const int* __restrict__ xlen,
                                  const int* __restrict__ ylen,
                                  float* __restrict__ out, int B, int Tp, int U1) {
  __shared__ float ll[32];
  int b = threadIdx.x;
  int U = U1 - 1;
  if (b < B) {
    int xmax = 0;
    for (int i = 0; i < B; ++i) xmax = max(xmax, xlen[i]);
    float scale = ceilf((float)xmax / (float)Tp);
    int xl = (int)ceilf((float)xlen[b] / scale);
    int tl = xl - 1;
    int yl = ylen[b];
    const float* lpb = lp_blank + (size_t)b * Tp * U1;
    const float* lpy = lp_y + (size_t)b * Tp * U;
    float alpha[128];  // U1 <= 128
    alpha[0] = 0.0f;
    for (int u = 1; u <= U; ++u) alpha[u] = alpha[u - 1] + lpy[u - 1];
    float llb = alpha[yl] + lpb[(size_t)tl * U1 + yl];  // valid when tl == 0
    for (int t = 1; t < Tp; ++t) {
      float a0 = alpha[0] + lpb[(size_t)(t - 1) * U1 + 0];
      float run = a0;
      for (int u = 1; u <= U; ++u) {
        float A_u = alpha[u] + lpb[(size_t)(t - 1) * U1 + u];
        run = logaddexp_f(A_u, run + lpy[(size_t)t * U + (u - 1)]);
        alpha[u] = run;
      }
      alpha[0] = a0;
      if (t == tl) llb = alpha[yl] + lpb[(size_t)tl * U1 + yl];
    }
    ll[b] = llb;
  }
  __syncthreads();
  if (threadIdx.x == 0) {
    float s = 0.0f;
    for (int i = 0; i < B; ++i) s += ll[i];
    out[0] = -s / (float)B;
  }
}

// ---------------------------------------------------------------------------
// Host orchestration
// ---------------------------------------------------------------------------
extern "C" void kernel_launch(void* const* d_in, const int* in_sizes, int n_in,
                              void* d_out, int out_size, void* d_ws, size_t ws_size,
                              hipStream_t stream) {
  (void)in_sizes; (void)out_size; (void)ws_size;
  const int Bsz = 4, T = 400, F_IN = 240, U = 60, Vv = 1024;
  const int ENC_H = 1024, ENC_PROJ = 640;
  const int DEC_EMB = 512, DEC_H = 512, DEC_PROJ = 640;
  const int JH = 512;
  const int U1 = U + 1;
  if (n_in < 43) return;

  // --- input pointers (setup_inputs dict order, params flattened in insertion order)
  const float* xs   = (const float*)d_in[0];
  const int*   ys   = (const int*)d_in[1];
  const int*   xlen = (const int*)d_in[2];
  const int*   ylen = (const int*)d_in[3];
  int p = 4;
  const float* in_ln_g = (const float*)d_in[p++];
  const float* in_ln_b = (const float*)d_in[p++];
  const float *eWih[4], *eWhh[4], *eBih[4], *eBhh[4], *eLng[4], *eLnb[4];
  for (int l = 0; l < 4; ++l) {
    eWih[l] = (const float*)d_in[p++];
    eWhh[l] = (const float*)d_in[p++];
    eBih[l] = (const float*)d_in[p++];
    eBhh[l] = (const float*)d_in[p++];
    eLng[l] = (const float*)d_in[p++];
    eLnb[l] = (const float*)d_in[p++];
  }
  const float* enc_proj_W = (const float*)d_in[p++];
  const float* enc_proj_b = (const float*)d_in[p++];
  const float* dec_embed  = (const float*)d_in[p++];
  const float* dWih = (const float*)d_in[p++];
  const float* dWhh = (const float*)d_in[p++];
  const float* dBih = (const float*)d_in[p++];
  const float* dBhh = (const float*)d_in[p++];
  const float* dec_proj_W = (const float*)d_in[p++];
  const float* dec_proj_b = (const float*)d_in[p++];
  const float* joint_W1 = (const float*)d_in[p++];
  const float* joint_b1 = (const float*)d_in[p++];
  const float* joint_W2 = (const float*)d_in[p++];
  const float* joint_b2 = (const float*)d_in[p++];

  // --- workspace bump allocator
  char* wsp = (char*)d_ws;
  size_t off = 0;
  auto alloc = [&](size_t bytes) -> void* {
    off = (off + 255) & ~(size_t)255;
    void* q = wsp + off;
    off += bytes;
    return q;
  };

  // bf16 weights
  bf16_t* wih_b[4]; bf16_t* whh_b[4];
  size_t wihElems[4] = {(size_t)4 * ENC_H * F_IN, (size_t)4 * ENC_H * ENC_H,
                        (size_t)4 * ENC_H * ENC_H, (size_t)4 * ENC_H * ENC_H};
  for (int l = 0; l < 4; ++l) wih_b[l] = (bf16_t*)alloc(wihElems[l] * 2);
  for (int l = 0; l < 4; ++l) whh_b[l] = (bf16_t*)alloc((size_t)4 * ENC_H * ENC_H * 2);
  bf16_t* dwih_b = (bf16_t*)alloc((size_t)4 * DEC_H * DEC_EMB * 2);
  bf16_t* dwhh_b = (bf16_t*)alloc((size_t)4 * DEC_H * DEC_H * 2);
  bf16_t* epW_b  = (bf16_t*)alloc((size_t)ENC_PROJ * ENC_H * 2);
  bf16_t* dpW_b  = (bf16_t*)alloc((size_t)DEC_PROJ * DEC_H * 2);
  bf16_t* jW1_b  = (bf16_t*)alloc((size_t)JH * (ENC_PROJ + DEC_PROJ) * 2);
  bf16_t* jW2_b  = (bf16_t*)alloc((size_t)Vv * JH * 2);

  // activations
  size_t actE = (size_t)Bsz * T * ENC_H;
  float*  Pf[2] = {(float*)alloc(actE * 4), (float*)alloc(actE * 4)};
  bf16_t* Pb[2] = {(bf16_t*)alloc(actE * 2), (bf16_t*)alloc(actE * 2)};
  float*  hsF = (float*)alloc(actE * 4);
  bf16_t* hsB = (bf16_t*)alloc(actE * 2);
  float*  gates = (float*)alloc((size_t)Bsz * T * 4 * ENC_H * 4);
  bf16_t* hbuf0 = (bf16_t*)alloc((size_t)16 * ENC_H * 2);
  bf16_t* hbuf1 = (bf16_t*)alloc((size_t)16 * ENC_H * 2);
  float*  cbuf  = (float*)alloc((size_t)4 * ENC_H * 4);
  bf16_t* henc_b = (bf16_t*)alloc((size_t)Bsz * (T / 2) * ENC_PROJ * 2);
  bf16_t* emb_b  = (bf16_t*)alloc((size_t)Bsz * U1 * DEC_EMB * 2);
  bf16_t* hdec_b = (bf16_t*)alloc((size_t)Bsz * U1 * DEC_PROJ * 2);
  float*  pe_f = (float*)alloc((size_t)Bsz * (T / 2) * JH * 4);
  float*  pd_f = (float*)alloc((size_t)Bsz * U1 * JH * 4);
  float*  lp_blank = (float*)alloc((size_t)Bsz * (T / 2) * U1 * 4);
  float*  lp_y     = (float*)alloc((size_t)Bsz * (T / 2) * U * 4);

  auto conv = [&](const float* src, bf16_t* dst, size_t n) {
    f32_to_bf16_kernel<<<(unsigned)((n + 255) / 256), 256, 0, stream>>>(src, dst, n);
  };
  auto gemm = [&](const bf16_t* A, int lda, const bf16_t* W, int ldw,
                  const float* b0, const float* b1v, float* Cf, bf16_t* Cb, int ldc,
                  int M, int N, int K) {
    int tiles = ((M + 15) / 16) * ((N + 15) / 16);
    unsigned grid = (unsigned)((tiles + 7) / 8);
    bool guard = (M & 15) || (N & 15) || (K & 31);
    if (guard)
      gemm_bf16_wmma<true><<<grid, 256, 0, stream>>>(A, lda, W, ldw, b0, b1v,
                                                     Cf, Cb, ldc, M, N, K);
    else
      gemm_bf16_wmma<false><<<grid, 256, 0, stream>>>(A, lda, W, ldw, b0, b1v,
                                                      Cf, Cb, ldc, M, N, K);
  };

  // --- weight conversions
  { int d = F_IN;
    for (int l = 0; l < 4; ++l) {
      conv(eWih[l], wih_b[l], (size_t)4 * ENC_H * d);
      conv(eWhh[l], whh_b[l], (size_t)4 * ENC_H * ENC_H);
      d = ENC_H;
    } }
  conv(dWih, dwih_b, (size_t)4 * DEC_H * DEC_EMB);
  conv(dWhh, dwhh_b, (size_t)4 * DEC_H * DEC_H);
  conv(enc_proj_W, epW_b, (size_t)ENC_PROJ * ENC_H);
  conv(dec_proj_W, dpW_b, (size_t)DEC_PROJ * DEC_H);
  conv(joint_W1, jW1_b, (size_t)JH * (ENC_PROJ + DEC_PROJ));
  conv(joint_W2, jW2_b, (size_t)Vv * JH);

  // --- encoder -------------------------------------------------------------
  layernorm_kernel<<<Bsz * T, 256, 0, stream>>>(xs, in_ln_g, in_ln_b, Pf[0], Pb[0],
                                                Bsz * T, F_IN);
  int pi = 0, d = F_IN, Tcur = T;
  for (int l = 0; l < 4; ++l) {
    int M = Bsz * Tcur;
    gemm(Pb[pi], d, wih_b[l], d, eBih[l], eBhh[l], gates, nullptr, 4 * ENC_H,
         M, 4 * ENC_H, d);
    zero_b16_kernel<<<(16 * ENC_H + 255) / 256, 256, 0, stream>>>(hbuf0, (size_t)16 * ENC_H);
    zero_b16_kernel<<<(16 * ENC_H + 255) / 256, 256, 0, stream>>>(hbuf1, (size_t)16 * ENC_H);
    zero_f32_kernel<<<(4 * ENC_H + 255) / 256, 256, 0, stream>>>(cbuf, (size_t)4 * ENC_H);
    for (int t = 0; t < Tcur; ++t) {
      bf16_t* hp = (t & 1) ? hbuf1 : hbuf0;
      bf16_t* hn = (t & 1) ? hbuf0 : hbuf1;
      lstm_step_kernel<<<ENC_H / 16, 128, 0, stream>>>(hp, hn, whh_b[l], gates, cbuf,
                                                       hsF, hsB, ENC_H, Tcur, t);
    }
    if (l > 0) {
      size_t n = (size_t)M * ENC_H;
      add_inplace_kernel<<<(unsigned)((n + 255) / 256), 256, 0, stream>>>(hsF, Pf[pi], n);
    }
    layernorm_kernel<<<M, 256, 0, stream>>>(hsF, eLng[l], eLnb[l], Pf[1 - pi], Pb[1 - pi],
                                            M, ENC_H);
    if (l == 1) {
      size_t n = (size_t)Bsz * (Tcur / 2) * ENC_H;
      time_reduce_kernel<<<(unsigned)((n + 255) / 256), 256, 0, stream>>>(
          Pf[1 - pi], Pf[pi], Pb[pi], Bsz, Tcur, ENC_H);
      Tcur /= 2;  // cur stays Pf[pi]
    } else {
      pi ^= 1;
    }
    d = ENC_H;
  }
  int Tp = Tcur;            // 200
  int Me = Bsz * Tp;        // 800
  gemm(Pb[pi], ENC_H, epW_b, ENC_H, enc_proj_b, nullptr, nullptr, henc_b, ENC_PROJ,
       Me, ENC_PROJ, ENC_H);

  // --- decoder -------------------------------------------------------------
  int Md = Bsz * U1;  // 244
  {
    size_t n = (size_t)Md * DEC_EMB;
    embed_kernel<<<(unsigned)((n + 255) / 256), 256, 0, stream>>>(ys, dec_embed, emb_b,
                                                                  Bsz, U1, DEC_EMB);
  }
  gemm(emb_b, DEC_EMB, dwih_b, DEC_EMB, dBih, dBhh, gates, nullptr, 4 * DEC_H,
       Md, 4 * DEC_H, DEC_EMB);
  zero_b16_kernel<<<(16 * DEC_H + 255) / 256, 256, 0, stream>>>(hbuf0, (size_t)16 * DEC_H);
  zero_b16_kernel<<<(16 * DEC_H + 255) / 256, 256, 0, stream>>>(hbuf1, (size_t)16 * DEC_H);
  zero_f32_kernel<<<(4 * DEC_H + 255) / 256, 256, 0, stream>>>(cbuf, (size_t)4 * DEC_H);
  for (int t = 0; t < U1; ++t) {
    bf16_t* hp = (t & 1) ? hbuf1 : hbuf0;
    bf16_t* hn = (t & 1) ? hbuf0 : hbuf1;
    lstm_step_kernel<<<DEC_H / 16, 128, 0, stream>>>(hp, hn, dwhh_b, gates, cbuf,
                                                     hsF, hsB, DEC_H, U1, t);
  }
  gemm(hsB, DEC_H, dpW_b, DEC_H, dec_proj_b, nullptr, nullptr, hdec_b, DEC_PROJ,
       Md, DEC_PROJ, DEC_H);

  // --- joint ---------------------------------------------------------------
  gemm(henc_b, ENC_PROJ, jW1_b, ENC_PROJ + DEC_PROJ, nullptr, nullptr, pe_f, nullptr,
       JH, Me, JH, ENC_PROJ);
  gemm(hdec_b, DEC_PROJ, jW1_b + ENC_PROJ, ENC_PROJ + DEC_PROJ, nullptr, nullptr, pd_f,
       nullptr, JH, Md, JH, DEC_PROJ);

  int R = Bsz * Tp * U1;
  size_t smem = (size_t)16 * JH * sizeof(bf16_t) + (size_t)16 * Vv * sizeof(float) +
                (256 + 32) * sizeof(float);
  joint_softmax_kernel<<<(R + 15) / 16, 256, smem, stream>>>(
      pe_f, pd_f, joint_b1, jW2_b, joint_b2, ys, ylen, lp_blank, lp_y,
      Bsz, Tp, U1, JH, Vv);

  rnnt_alpha_kernel<<<1, 32, 0, stream>>>(lp_blank, lp_y, xlen, ylen, (float*)d_out,
                                          Bsz, Tp, U1);
}
